// InputEmbedding_21844203667614
// MI455X (gfx1250) — matrix operand order repaired and checked by reading, not compiled, not run on hardware
//
#include <hip/hip_runtime.h>

#define DEVINL __device__ __forceinline__
#define USE_ASYNC_LDS 1

typedef __attribute__((ext_vector_type(16))) _Float16 v16h;
typedef __attribute__((ext_vector_type(8)))  _Float16 v8h;
typedef __attribute__((ext_vector_type(8)))  float    v8f;

static constexpr int BB  = 8;
static constexpr int NN  = 2048;
static constexpr int KNB = 20;       // k neighbors (reference k=20)

DEVINL float lrelu(float x) { return x >= 0.f ? x : 0.2f * x; }

// ---- WMMA fragment builders (A and B both use contiguous k = 16*g + s) ----
// A 16x32 f16 from row-major W[O][K]: per lane, halves are two contiguous
// 8-half runs -> two 16B LDS loads + shuffle merge.
DEVINL v16h fragA(const _Float16* W, int o0, int k0, int ld, int lane) {
  int m = lane & 15, g = (lane >> 4) & 1;
  const _Float16* p = W + (o0 + m) * ld + k0 + 16 * g;
  v8h lo = *(const v8h*)p;
  v8h hi = *(const v8h*)(p + 8);
  return __builtin_shufflevector(lo, hi, 0, 1, 2, 3, 4, 5, 6, 7,
                                 8, 9, 10, 11, 12, 13, 14, 15);
}
// B 32x16 f16 from point-major F[P][K]: per lane halves are k = k0+16g+h,
// one aligned 32-byte run -> single v16h LDS load (2x ds_load_b128).
DEVINL v16h fragB(const _Float16* F, int p0, int k0, int ld, int lane) {
  int n = lane & 15, g = (lane >> 4) & 1;
  return *(const v16h*)(F + (p0 + n) * ld + k0 + 16 * g);
}
DEVINL v8f wmma32(v16h a, v16h b, v8f c) {
  return __builtin_amdgcn_wmma_f32_16x16x32_f16(false, a, false, b, (short)0, c, false, false);
}

// full-wave cross-lane argmax step (tie-break: lowest index)
DEVINL void argmax_sh(float& v, int& i, int off) {
  float vo = __shfl_xor(v, off, 32);
  int   io = __shfl_xor(i, off, 32);
  if (vo > v || (vo == v && io < i)) { v = vo; i = io; }
}

// =====================================================================
// KNN: one workgroup per (b,n) row; LDS distance row + 20 argmax rounds.
// pd[m] = -||x_n - x_m||^2.  For C==3 the whole point block (24KB) is
// staged into LDS via async load-to-LDS (CDNA5 ASYNCcnt path).
// =====================================================================
template <int C>
__global__ __launch_bounds__(256) void knn_kernel(const float* __restrict__ pts,
                                                  int* __restrict__ idxo) {
  __shared__ float pd[NN];
  __shared__ __align__(16) float pl[(C == 3) ? C * NN : 1];
  __shared__ float rv[8];
  __shared__ int   ri[8];
  const int tid = threadIdx.x, wave = tid >> 5, lane = tid & 31;
  const int b = blockIdx.x >> 11;
  const int n = blockIdx.x & (NN - 1);

  if constexpr (C == 3) {
#if USE_ASYNC_LDS
    {
      const char* gsrc = (const char*)(pts + (size_t)b * C * NN);
      for (int j = tid; j < (C * NN * 4) / 16; j += 256) {
        unsigned ldsa = (unsigned)(uintptr_t)((char*)pl + (size_t)j * 16);
        const char* ga = gsrc + (size_t)j * 16;
        asm volatile("global_load_async_to_lds_b128 %0, %1, off"
                     :: "v"(ldsa), "v"(ga) : "memory");
      }
      asm volatile("s_wait_asynccnt 0x0" ::: "memory");
    }
#else
    for (int j = tid; j < C * NN; j += 256) pl[j] = pts[(size_t)b * C * NN + j];
#endif
    __syncthreads();
    float c0 = pl[0 * NN + n], c1 = pl[1 * NN + n], c2 = pl[2 * NN + n];
    for (int m = tid; m < NN; m += 256) {
      float d0 = pl[0 * NN + m] - c0;
      float d1 = pl[1 * NN + m] - c1;
      float d2 = pl[2 * NN + m] - c2;
      pd[m] = -(d0 * d0 + fmaf(d1, d1, d2 * d2));
    }
  } else {
    __shared__ float cen[C];
    if (tid < C) cen[tid] = pts[((size_t)b * C + tid) * NN + n];
    __syncthreads();
    for (int m = tid; m < NN; m += 256) {
      float s = 0.f;
      for (int c = 0; c < C; ++c) {
        float d = pts[((size_t)b * C + c) * NN + m] - cen[c];
        s = fmaf(d, d, s);
      }
      pd[m] = -s;
    }
  }
  __syncthreads();

  for (int it = 0; it < KNB; ++it) {
    float bv = -3.0e38f; int bi = 0;
    for (int m = tid; m < NN; m += 256) {
      float v = pd[m];
      if (v > bv || (v == bv && m < bi)) { bv = v; bi = m; }
    }
    // wave-level shuffle argmax, then 8-way merge by thread 0
    argmax_sh(bv, bi, 16); argmax_sh(bv, bi, 8); argmax_sh(bv, bi, 4);
    argmax_sh(bv, bi, 2);  argmax_sh(bv, bi, 1);
    if (lane == 0) { rv[wave] = bv; ri[wave] = bi; }
    __syncthreads();
    if (tid == 0) {
      float v = rv[0]; int i = ri[0];
#pragma unroll
      for (int w = 1; w < 8; ++w)
        if (rv[w] > v || (rv[w] == v && ri[w] < i)) { v = rv[w]; i = ri[w]; }
      idxo[((size_t)b * NN + n) * KNB + it] = i;
      pd[i] = -3.4e38f;
    }
    __syncthreads();
  }
}

// =====================================================================
// Generic fused edge-conv WMMA kernel.
// PROD: 0 = edge6 -> preconv(64x6)+BNfold+leaky -> Fh[p][64]
//       1 = edge128 from 64-ch features -> Fh[p][128]
//       2 = edge128 -> WMMA W3(64x128)+BNfold+leaky -> Fh[p][64]
//       3 = raw edge6 zero-padded to 32 K-slots -> Fh[p][32]
// MODE: 0 = stats (sum/sumsq per out channel, LDS atomics + global atomics)
//       1 = BN fold + leaky after max over k, write out0/out1 [B,outC,N]
// Tile: 4 points x 20 neighbors = 80 cols = 5 N-tiles of 16.
// =====================================================================
template <int CIN, int COUT, int PROD, int MODE>
__global__ __launch_bounds__(256) void edgeconv_kernel(
    const float* __restrict__ pts, const int* __restrict__ idx,
    const float* __restrict__ Wpre, const float* __restrict__ apre, const float* __restrict__ cpre,
    const float* __restrict__ Wmain, const float* __restrict__ amain, const float* __restrict__ cmain,
    float* __restrict__ sums, float* __restrict__ sumsq,
    float* __restrict__ out0, int out0C, int out0co,
    float* __restrict__ out1, int out1C, int out1co) {
  constexpr int PC  = (PROD == 0 || PROD == 3) ? 3 : 64;  // point-feature channels
  constexpr int NP  = 80;                                  // points per tile
  constexpr int PT  = 5;                                   // 16-wide point tiles
  constexpr int TO  = COUT / 16;
  constexpr int TOT = TO * PT;
  constexpr int MAXT = (TOT + 7) / 8;

  constexpr size_t szW   = (size_t)COUT * CIN * 2;
  constexpr size_t szJn  = NP * 4;
  constexpr size_t szP   = (PROD == 0) ? (szJn + 64 * 6 * 4 + NP * 6 * 4 + 64 * 4 + 64 * 4)
                         : (PROD == 2) ? (szJn + (size_t)64 * 128 * 2 + (size_t)NP * 128 * 2)
                                       : szJn;
  constexpr size_t szOut = (MODE == 1) ? (size_t)64 * NP * 4 : 0;
  constexpr size_t szPO  = (szP > szOut) ? szP : szOut;
  constexpr size_t szF   = (size_t)NP * CIN * 2;
  constexpr size_t szS   = (MODE == 0) ? (size_t)COUT * 2 * 4 : 0;
  constexpr size_t SMEM  = szW + szPO + szF + szS;
  __shared__ __align__(32) unsigned char smem[SMEM];

  _Float16* Wl = (_Float16*)smem;                 // [COUT][CIN] row-major
  unsigned char* Pbase = smem + szW;
  int*      jn_l = (int*)Pbase;
  float*    OutL = (float*)Pbase;                 // MODE1, aliases producer scratch
  _Float16* Fh   = (_Float16*)(smem + szW + szPO);// [NP][CIN] point-major
  float*    Ssum = (float*)(smem + szW + szPO + szF);
  float*    Ssq  = Ssum + COUT;

  const int tid = threadIdx.x, wave = tid >> 5, lane = tid & 31, hi = (lane >> 4) & 1;
  const int b = blockIdx.x >> 9;                  // 512 tiles per batch (2048/4)
  const int n0 = (blockIdx.x & 511) * 4;

  // ---- preamble: weights, neighbor indices, zero stats ----
  if (tid < 16) __builtin_prefetch(Wmain + (size_t)tid * 16);
  if constexpr (PROD == 3) {
    for (int j = tid; j < COUT * 32; j += 256) {
      int o = j >> 5, c = j & 31;
      Wl[j] = (_Float16)((c < 6) ? Wmain[o * 6 + c] : 0.f);
    }
    for (int j = tid; j < NP * 32; j += 256) Fh[j] = (_Float16)0.f;  // zero-pad K
  } else {
    for (int j = tid; j < COUT * CIN; j += 256) Wl[j] = (_Float16)Wmain[j];
  }
  if (tid < NP) {
    int nl = tid / KNB, kk = tid % KNB;
    jn_l[tid] = idx[((size_t)b * NN + (n0 + nl)) * KNB + kk];
  }
  if constexpr (MODE == 0) {
    for (int j = tid; j < 2 * COUT; j += 256) Ssum[j] = 0.f;
  }
  float* wpre_l = nullptr; float* f6_l = nullptr; float* apre_l = nullptr; float* cpre_l = nullptr;
  _Float16* W3l = nullptr; _Float16* Ghl = nullptr;
  if constexpr (PROD == 0) {
    wpre_l = (float*)(Pbase + szJn);
    f6_l   = wpre_l + 64 * 6;
    apre_l = f6_l + NP * 6;
    cpre_l = apre_l + 64;
    for (int j = tid; j < 64 * 6; j += 256) wpre_l[j] = Wpre[j];
    if (tid < 64) { apre_l[tid] = apre[tid]; cpre_l[tid] = cpre[tid]; }
  }
  if constexpr (PROD == 2) {
    W3l = (_Float16*)(Pbase + szJn);   // [64][128]
    Ghl = W3l + 64 * 128;              // [NP][128] point-major
    for (int j = tid; j < 64 * 128; j += 256) W3l[j] = (_Float16)Wpre[j];
  }
  __syncthreads();

  // ---- producer: build Fh[NP][CIN] f16 ----
  if constexpr (PROD == 3) {
    if (tid < NP) {
      int nl = tid / KNB; int n = n0 + nl; int jn = jn_l[tid];
#pragma unroll
      for (int c = 0; c < 3; ++c) {
        float ctr = pts[((size_t)b * PC + c) * NN + n];
        float nb  = pts[((size_t)b * PC + c) * NN + jn];
        Fh[tid * 32 + c]     = (_Float16)(nb - ctr);
        Fh[tid * 32 + 3 + c] = (_Float16)ctr;
      }
    }
  }
  if constexpr (PROD == 0) {
    if (tid < NP) {
      int nl = tid / KNB; int n = n0 + nl; int jn = jn_l[tid];
#pragma unroll
      for (int c = 0; c < 3; ++c) {
        float ctr = pts[((size_t)b * PC + c) * NN + n];
        float nb  = pts[((size_t)b * PC + c) * NN + jn];
        f6_l[tid * 6 + c]     = nb - ctr;
        f6_l[tid * 6 + 3 + c] = ctr;
      }
    }
    __syncthreads();
    for (int jj = tid; jj < NP * 64; jj += 256) {
      int p = jj >> 6, o = jj & 63;
      float y = 0.f;
#pragma unroll
      for (int c = 0; c < 6; ++c) y = fmaf(wpre_l[o * 6 + c], f6_l[p * 6 + c], y);
      Fh[p * 64 + o] = (_Float16)lrelu(apre_l[o] * y + cpre_l[o]);
    }
  }
  if constexpr (PROD == 1 || PROD == 2) {
    _Float16* dst = (PROD == 1) ? Fh : Ghl;
    for (int jj = tid; jj < NP * 128; jj += 256) {
      int p = jj >> 7, c = jj & 127;
      int nl = p / KNB; int n = n0 + nl; int jn = jn_l[p];
      float v;
      if (c < 64) v = pts[((size_t)b * 64 + c) * NN + jn] - pts[((size_t)b * 64 + c) * NN + n];
      else        v = pts[((size_t)b * 64 + (c - 64)) * NN + n];
      dst[p * 128 + c] = (_Float16)v;
    }
  }
  if constexpr (PROD == 2) {
    __syncthreads();
    for (int t = wave; t < 20; t += 8) {   // 4 o-tiles x 5 p-tiles, K=128
      int ot = t / 5, pt = t % 5;
      v8f acc = {};
      for (int kc = 0; kc < 128; kc += 32)
        acc = wmma32(fragA(W3l, ot * 16, kc, 128, lane), fragB(Ghl, pt * 16, kc, 128, lane), acc);
      int pl = pt * 16 + (lane & 15);
#pragma unroll
      for (int r = 0; r < 8; ++r) {
        int o = ot * 16 + r + 8 * hi;
        Fh[pl * 64 + o] = (_Float16)lrelu(apre[o] * acc[r] + cpre[o]);
      }
    }
  }
  __syncthreads();

  // ---- main WMMA GEMM: COUT x 80 ----
  v8f accs[MAXT];
#pragma unroll
  for (int i = 0; i < MAXT; ++i) {
    int t = wave + i * 8;
    if (t < TOT) {
      int ot = t / PT, pt = t % PT;
      v8f acc = {};
      for (int kc = 0; kc < CIN; kc += 32)
        acc = wmma32(fragA(Wl, ot * 16, kc, CIN, lane), fragB(Fh, pt * 16, kc, CIN, lane), acc);
      if constexpr (MODE == 0) {
#pragma unroll
        for (int r = 0; r < 8; ++r) {
          int o = ot * 16 + r + 8 * hi;
          float v = acc[r];
          atomicAdd(&Ssum[o], v);
          atomicAdd(&Ssq[o], v * v);
        }
      } else {
        accs[i] = acc;
      }
    }
  }

  if constexpr (MODE == 0) {
    __syncthreads();
    for (int j = tid; j < COUT; j += 256) {
      atomicAdd(&sums[j], Ssum[j]);
      atomicAdd(&sumsq[j], Ssq[j]);
    }
  } else {
    for (int oh = 0; oh < COUT / 64; ++oh) {
      __syncthreads();
#pragma unroll
      for (int i = 0; i < MAXT; ++i) {
        int t = wave + i * 8;
        if (t < TOT) {
          int ot = t / PT, pt = t % PT;
          if ((ot >> 2) == oh) {
            int pl = pt * 16 + (lane & 15);
#pragma unroll
            for (int r = 0; r < 8; ++r) {
              int ol = (ot & 3) * 16 + r + 8 * hi;
              OutL[ol * NP + pl] = accs[i][r];
            }
          }
        }
      }
      __syncthreads();
      // max over k (monotone act -> fold after max), write outputs
      for (int j = tid; j < 64 * 4; j += 256) {
        int ol = j >> 2, nl = j & 3;
        int o = oh * 64 + ol, n = n0 + nl;
        float m = -3.4e38f;
#pragma unroll
        for (int kk = 0; kk < KNB; ++kk) m = fmaxf(m, OutL[ol * NP + nl * KNB + kk]);
        float v = lrelu(amain[o] * m + cmain[o]);
        out0[((size_t)b * out0C + (o + out0co)) * NN + n] = v;
        if (out1) out1[((size_t)b * out1C + (o + out1co)) * NN + n] = v;
      }
    }
    __syncthreads();
  }
}

// =====================================================================
// STN conv1: tw3 [1024,128] over hm [B,128,N].
// MODE 0: stats per channel.  MODE 1: BN fold + leaky after max over N.
// Block handles 128 out channels; 48-point tiles (3 x 16), point-major LDS.
// =====================================================================
template <int MODE>
__global__ __launch_bounds__(256) void conv1_kernel(
    const float* __restrict__ hm, const float* __restrict__ W,
    const float* __restrict__ aa, const float* __restrict__ cc,
    float* __restrict__ sums, float* __restrict__ sumsq, float* __restrict__ gout) {
  constexpr int NP = 48;
  constexpr int NBLK = 43;  // ceil(2048/48)
  __shared__ __align__(32) _Float16 Wl[128 * 128];  // [O][K]
  __shared__ __align__(32) _Float16 Fl[NP * 128];   // [P][K] point-major
  __shared__ float OutL[64 * NP];
  __shared__ float Sred[256];
  const int tid = threadIdx.x, wave = tid >> 5, lane = tid & 31, hi = (lane >> 4) & 1;
  int b, ob, nb0, nbN;
  if constexpr (MODE == 0) {
    int bid = blockIdx.x;
    b = bid / (8 * NBLK); ob = (bid / NBLK) % 8; nb0 = bid % NBLK; nbN = nb0 + 1;
  } else {
    b = blockIdx.x / 8; ob = blockIdx.x % 8; nb0 = 0; nbN = NBLK;
  }
  for (int j = tid; j < 128 * 128; j += 256) Wl[j] = (_Float16)W[(size_t)ob * 128 * 128 + j];
  if constexpr (MODE == 0) { Sred[tid] = 0.f; }
  else { if (tid < 128) Sred[tid] = -3.4e38f; }
  __syncthreads();

  for (int nb = nb0; nb < nbN; ++nb) {
    int n0 = nb * NP;
    // prefetch next tile's rows while computing this one
    if (MODE == 1 && nb + 1 < nbN && tid < 128)
      __builtin_prefetch(&hm[((size_t)b * 128 + tid) * NN + (nb + 1) * NP]);
    for (int j = tid; j < 128 * NP; j += 256) {
      int c = j / NP, p = j % NP; int n = n0 + p;   // p-fast: coalesced global reads
      Fl[p * 128 + c] = (_Float16)((n < NN) ? hm[((size_t)b * 128 + c) * NN + n] : 0.f);
    }
    __syncthreads();
    v8f accs[3];
#pragma unroll
    for (int i = 0; i < 3; ++i) {
      int t = wave + i * 8;          // 24 tiles: 8 o-tiles x 3 p-tiles
      int ot = t / 3, pt = t % 3;
      v8f acc = {};
      for (int kc = 0; kc < 128; kc += 32)
        acc = wmma32(fragA(Wl, ot * 16, kc, 128, lane), fragB(Fl, pt * 16, kc, 128, lane), acc);
      if constexpr (MODE == 0) {
        int pl = pt * 16 + (lane & 15);
        bool pv = (n0 + pl) < NN;
#pragma unroll
        for (int r = 0; r < 8; ++r) {
          int o = ot * 16 + r + 8 * hi;
          float v = pv ? acc[r] : 0.f;
          atomicAdd(&Sred[o], v);
          atomicAdd(&Sred[128 + o], v * v);
        }
      } else {
        accs[i] = acc;
      }
    }
    if constexpr (MODE == 1) {
      for (int oh = 0; oh < 2; ++oh) {
        __syncthreads();
#pragma unroll
        for (int i = 0; i < 3; ++i) {
          int t = wave + i * 8;
          int ot = t / 3, pt = t % 3;
          if ((ot >> 2) == oh) {
            int pl = pt * 16 + (lane & 15);
#pragma unroll
            for (int r = 0; r < 8; ++r) {
              int ol = (ot & 3) * 16 + r + 8 * hi;
              OutL[ol * NP + pl] = accs[i][r];
            }
          }
        }
        __syncthreads();
        if (tid < 64) {
          float m = Sred[oh * 64 + tid];
          int lim = (NN - n0 < NP) ? (NN - n0) : NP;
          for (int p = 0; p < lim; ++p) m = fmaxf(m, OutL[tid * NP + p]);
          Sred[oh * 64 + tid] = m;
        }
      }
    }
    __syncthreads();
  }
  if constexpr (MODE == 0) {
    for (int j = tid; j < 128; j += 256) {
      atomicAdd(&sums[ob * 128 + j], Sred[j]);
      atomicAdd(&sumsq[ob * 128 + j], Sred[128 + j]);
    }
  } else {
    if (tid < 128) {
      int o = ob * 128 + tid;
      gout[(size_t)b * 1024 + o] = lrelu(aa[o] * Sred[tid] + cc[o]);
    }
  }
}

// =====================================================================
// FC head (single workgroup; B=8)
// =====================================================================
__global__ __launch_bounds__(256) void fc_kernel(
    const float* __restrict__ g, const float* __restrict__ lin1, const float* __restrict__ bn4,
    const float* __restrict__ lin2, const float* __restrict__ bn5,
    const float* __restrict__ ttw, const float* __restrict__ ttb,
    float* __restrict__ tbuf, float* __restrict__ outt) {
  __shared__ float z1[8 * 512];
  __shared__ float z2[8 * 256];
  int tid = threadIdx.x;
  for (int j = tid; j < 8 * 512; j += 256) {
    int o = j >> 3, b = j & 7;
    float s = 0.f;
    const float* w = lin1 + (size_t)o * 1024;
    const float* gg = g + (size_t)b * 1024;
    for (int c = 0; c < 1024; ++c) s = fmaf(w[c], gg[c], s);
    z1[b * 512 + o] = s;
  }
  __syncthreads();
  for (int o = tid; o < 512; o += 256) {
    float mu = 0.f; for (int b = 0; b < 8; ++b) mu += z1[b * 512 + o]; mu *= 0.125f;
    float vr = 0.f; for (int b = 0; b < 8; ++b) { float d = z1[b * 512 + o] - mu; vr = fmaf(d, d, vr); } vr *= 0.125f;
    float a = bn4[o] * rsqrtf(vr + 1e-5f);
    float cb = bn4[512 + o] - mu * a;
    for (int b = 0; b < 8; ++b) z1[b * 512 + o] = lrelu(a * z1[b * 512 + o] + cb);
  }
  __syncthreads();
  for (int j = tid; j < 8 * 256; j += 256) {
    int o = j >> 3, b = j & 7;
    float s = 0.f;
    for (int c = 0; c < 512; ++c) s = fmaf(lin2[(size_t)o * 512 + c], z1[b * 512 + c], s);
    z2[b * 256 + o] = s;
  }
  __syncthreads();
  if (tid < 256) {
    int o = tid;
    float mu = 0.f; for (int b = 0; b < 8; ++b) mu += z2[b * 256 + o]; mu *= 0.125f;
    float vr = 0.f; for (int b = 0; b < 8; ++b) { float d = z2[b * 256 + o] - mu; vr = fmaf(d, d, vr); } vr *= 0.125f;
    float a = bn5[o] * rsqrtf(vr + 1e-5f);
    float cb = bn5[256 + o] - mu * a;
    for (int b = 0; b < 8; ++b) z2[b * 256 + o] = lrelu(a * z2[b * 256 + o] + cb);
  }
  __syncthreads();
  if (tid < 72) {
    int b = tid / 9, o = tid % 9;
    float s = ttb[o];
    for (int c = 0; c < 256; ++c) s = fmaf(ttw[o * 256 + c], z2[b * 256 + c], s);
    tbuf[b * 9 + o] = s;
    outt[b * 9 + o] = s;
  }
}

// xt2[b,d,n] = sum_c x[b,c,n] * t[b,c,d]
__global__ void transform_kernel(const float* __restrict__ x, const float* __restrict__ t,
                                 float* __restrict__ xo) {
  int i = blockIdx.x * 256 + threadIdx.x;
  if (i >= BB * 3 * NN) return;
  int n = i & (NN - 1);
  int d = (i >> 11) % 3;
  int b = i / (3 * NN);
  float s = 0.f;
#pragma unroll
  for (int c = 0; c < 3; ++c) s = fmaf(x[((size_t)b * 3 + c) * NN + n], t[b * 9 + c * 3 + d], s);
  xo[((size_t)b * 3 + d) * NN + n] = s;
}

__global__ void zero_kernel(float* __restrict__ p, int n) {
  int i = blockIdx.x * 256 + threadIdx.x;
  if (i < n) p[i] = 0.f;
}

// fold BN: a = g*rsqrt(var+eps), c = beta - mean*a
__global__ void finalize_bn(const float* __restrict__ sums, const float* __restrict__ sumsq,
                            const float* __restrict__ gb, int C, float inv_cnt,
                            float* __restrict__ a, float* __restrict__ cb) {
  int t = blockIdx.x * 256 + threadIdx.x;
  if (t < C) {
    float mu = sums[t] * inv_cnt;
    float vr = sumsq[t] * inv_cnt - mu * mu;
    float aa = gb[t] * rsqrtf(vr + 1e-5f);
    a[t] = aa;
    cb[t] = gb[C + t] - mu * aa;
  }
}

// =====================================================================
extern "C" void kernel_launch(void* const* d_in, const int* in_sizes, int n_in,
                              void* d_out, int out_size, void* d_ws, size_t ws_size,
                              hipStream_t stream) {
  (void)in_sizes; (void)n_in; (void)out_size; (void)ws_size;
  const float* x    = (const float*)d_in[0];
  const float* tw1  = (const float*)d_in[1];
  const float* tbn1 = (const float*)d_in[2];
  const float* tw2  = (const float*)d_in[3];
  const float* tbn2 = (const float*)d_in[4];
  const float* tw3  = (const float*)d_in[5];
  const float* tbn3 = (const float*)d_in[6];
  const float* tl1  = (const float*)d_in[7];
  const float* tbn4 = (const float*)d_in[8];
  const float* tl2  = (const float*)d_in[9];
  const float* tbn5 = (const float*)d_in[10];
  const float* ttw  = (const float*)d_in[11];
  const float* ttb  = (const float*)d_in[12];
  const float* w1   = (const float*)d_in[13];
  const float* bn1  = (const float*)d_in[14];
  const float* w2   = (const float*)d_in[15];
  const float* bn2  = (const float*)d_in[16];
  const float* w3   = (const float*)d_in[17];
  const float* bn3  = (const float*)d_in[18];
  const float* w4   = (const float*)d_in[19];
  const float* bn4g = (const float*)d_in[20];
  float* out = (float*)d_out;

  // ---- workspace layout (< 16 MB) ----
  char* wp = (char*)d_ws;
  auto alloc = [&](size_t bytes) -> void* {
    void* p = (void*)wp;
    wp += (bytes + 255) & ~(size_t)255;
    return p;
  };
  int*   idxb  = (int*)alloc((size_t)BB * NN * KNB * 4);
  float* xt2   = (float*)alloc((size_t)BB * 3 * NN * 4);
  float* x1    = (float*)alloc((size_t)BB * 64 * NN * 4);
  float* hm    = (float*)alloc((size_t)BB * 128 * NN * 4);
  float* gbuf  = (float*)alloc((size_t)BB * 1024 * 4);
  float* tbuf  = (float*)alloc(72 * 4);
  float* sums  = (float*)alloc(1024 * 4);
  float* sumsq = (float*)alloc(1024 * 4);
  float* ab    = (float*)alloc((size_t)2 * (64 + 128 + 1024 + 64 + 64 + 64 + 64) * 4);
  float* aS1 = ab;          float* cS1 = aS1 + 64;
  float* aS2 = cS1 + 64;    float* cS2 = aS2 + 128;
  float* aS3 = cS2 + 128;   float* cS3 = aS3 + 1024;
  float* aM1 = cS3 + 1024;  float* cM1 = aM1 + 64;
  float* aM2 = cM1 + 64;    float* cM2 = aM2 + 64;
  float* aM3 = cM2 + 64;    float* cM3 = aM3 + 64;
  float* aM4 = cM3 + 64;    float* cM4 = aM4 + 64;

  const float inv_ek = 1.f / ((float)BB * NN * KNB);  // edge-conv BN element count
  const float inv_n  = 1.f / ((float)BB * NN);        // conv1 BN element count
  const int EC_GRID = BB * (NN / 4);                  // 4096 blocks
  auto zero2 = [&]() {
    zero_kernel<<<4, 256, 0, stream>>>(sums, 1024);
    zero_kernel<<<4, 256, 0, stream>>>(sumsq, 1024);
  };

  // ============ STN branch ============
  knn_kernel<3><<<BB * NN, 256, 0, stream>>>(x, idxb);

  zero2();
  edgeconv_kernel<32, 64, 3, 0><<<EC_GRID, 256, 0, stream>>>(
      x, idxb, nullptr, nullptr, nullptr, tw1, nullptr, nullptr,
      sums, sumsq, nullptr, 0, 0, nullptr, 0, 0);
  finalize_bn<<<1, 256, 0, stream>>>(sums, sumsq, tbn1, 64, inv_ek, aS1, cS1);

  zero2();
  edgeconv_kernel<64, 128, 0, 0><<<EC_GRID, 256, 0, stream>>>(
      x, idxb, tw1, aS1, cS1, tw2, nullptr, nullptr,
      sums, sumsq, nullptr, 0, 0, nullptr, 0, 0);
  finalize_bn<<<1, 256, 0, stream>>>(sums, sumsq, tbn2, 128, inv_ek, aS2, cS2);

  edgeconv_kernel<64, 128, 0, 1><<<EC_GRID, 256, 0, stream>>>(
      x, idxb, tw1, aS1, cS1, tw2, aS2, cS2,
      nullptr, nullptr, hm, 128, 0, nullptr, 0, 0);

  zero2();
  conv1_kernel<0><<<BB * 8 * 43, 256, 0, stream>>>(hm, tw3, nullptr, nullptr, sums, sumsq, nullptr);
  finalize_bn<<<4, 256, 0, stream>>>(sums, sumsq, tbn3, 1024, inv_n, aS3, cS3);
  conv1_kernel<1><<<BB * 8, 256, 0, stream>>>(hm, tw3, aS3, cS3, nullptr, nullptr, gbuf);

  fc_kernel<<<1, 256, 0, stream>>>(gbuf, tl1, tbn4, tl2, tbn5, ttw, ttb,
                                   tbuf, out + (size_t)BB * 128 * NN);
  transform_kernel<<<(BB * 3 * NN + 255) / 256, 256, 0, stream>>>(x, tbuf, xt2);

  // ============ Main branch, block 1 (3ch -> x1) ============
  knn_kernel<3><<<BB * NN, 256, 0, stream>>>(xt2, idxb);

  zero2();
  edgeconv_kernel<32, 64, 3, 0><<<EC_GRID, 256, 0, stream>>>(
      xt2, idxb, nullptr, nullptr, nullptr, w1, nullptr, nullptr,
      sums, sumsq, nullptr, 0, 0, nullptr, 0, 0);
  finalize_bn<<<1, 256, 0, stream>>>(sums, sumsq, bn1, 64, inv_ek, aM1, cM1);

  zero2();
  edgeconv_kernel<64, 64, 0, 0><<<EC_GRID, 256, 0, stream>>>(
      xt2, idxb, w1, aM1, cM1, w2, nullptr, nullptr,
      sums, sumsq, nullptr, 0, 0, nullptr, 0, 0);
  finalize_bn<<<1, 256, 0, stream>>>(sums, sumsq, bn2, 64, inv_ek, aM2, cM2);

  edgeconv_kernel<64, 64, 0, 1><<<EC_GRID, 256, 0, stream>>>(
      xt2, idxb, w1, aM1, cM1, w2, aM2, cM2,
      nullptr, nullptr, x1, 64, 0, out, 128, 0);

  // ============ Main branch, block 2 (64ch -> x2) ============
  knn_kernel<64><<<BB * NN, 256, 0, stream>>>(x1, idxb);

  zero2();
  edgeconv_kernel<128, 64, 1, 0><<<EC_GRID, 256, 0, stream>>>(
      x1, idxb, nullptr, nullptr, nullptr, w3, nullptr, nullptr,
      sums, sumsq, nullptr, 0, 0, nullptr, 0, 0);
  finalize_bn<<<1, 256, 0, stream>>>(sums, sumsq, bn3, 64, inv_ek, aM3, cM3);

  zero2();
  edgeconv_kernel<64, 64, 2, 0><<<EC_GRID, 256, 0, stream>>>(
      x1, idxb, w3, aM3, cM3, w4, nullptr, nullptr,
      sums, sumsq, nullptr, 0, 0, nullptr, 0, 0);
  finalize_bn<<<1, 256, 0, stream>>>(sums, sumsq, bn4g, 64, inv_ek, aM4, cM4);

  edgeconv_kernel<64, 64, 2, 1><<<EC_GRID, 256, 0, stream>>>(
      x1, idxb, w3, aM3, cM3, w4, aM4, cM4,
      nullptr, nullptr, out, 128, 64, nullptr, 0, 0);
}
